// VQ_VAE_65034394796142
// MI455X (gfx1250) — matrix operand order repaired and checked by reading, not compiled, not run on hardware
//
#include <hip/hip_runtime.h>

// ---------------------------------------------------------------------------
// VQ-VAE forward for MI455X (gfx1250, wave32, WMMA).
// All four GEMMs: v_wmma_f32_16x16x32_bf16, f32 accumulate, fused bias+act.
// Global->LDS staging uses CDNA5 async-to-LDS (ASYNCcnt) with double-buffered
// LDS so HBM/L2 latency overlaps the WMMA stream. N/K=400 zero-padded to 448.
// Grid is (Ntiles, Mtiles) so blocks sharing an A panel run consecutively and
// the panel stays L2-resident (x alone is 201 MB > 192 MB L2).
// ---------------------------------------------------------------------------

typedef __attribute__((ext_vector_type(4)))  int    v4i;
typedef __attribute__((ext_vector_type(8)))  float  v8f;
typedef __attribute__((ext_vector_type(8)))  __bf16 v8bf;
typedef __attribute__((ext_vector_type(16))) __bf16 v16bf;

#define GLOBAL_AS __attribute__((address_space(1)))
#define LDS_AS    __attribute__((address_space(3)))

#define LDT 72  // LDS tile row stride in halves: 64 data + 8 pad (144B, 16B-aligned)

static __device__ __forceinline__ unsigned short f2bf(float f) {
  // round-to-nearest-even f32 -> bf16 (scalar)
  unsigned u = __float_as_uint(f);
  unsigned r = u + 0x7FFFu + ((u >> 16) & 1u);
  return (unsigned short)(r >> 16);
}

// Pack two f32 into one dword of bf16x2 (RNE). Prefers the HW pack cvt.
static __device__ __forceinline__ unsigned packbf2(float a, float b) {
#if __has_builtin(__builtin_amdgcn_cvt_pk_bf16_f32)
  return __builtin_bit_cast(unsigned, __builtin_amdgcn_cvt_pk_bf16_f32(a, b));
#else
  unsigned ua = __float_as_uint(a);
  unsigned ub = __float_as_uint(b);
  ua = ua + 0x7FFFu + ((ua >> 16) & 1u);
  ub = ub + 0x7FFFu + ((ub >> 16) & 1u);
  return (ua >> 16) | (ub & 0xFFFF0000u);
#endif
}

// 16B async copy global -> LDS (ASYNCcnt-tracked, no VGPR staging).
// clang prototype (from probe diagnostic): (v4i AS1*, v4i AS3*, imm, imm)
static __device__ __forceinline__ void async_b128(const void* g, void* l) {
#if __has_builtin(__builtin_amdgcn_global_load_async_to_lds_b128)
  __builtin_amdgcn_global_load_async_to_lds_b128(
      (GLOBAL_AS v4i*)g, (LDS_AS v4i*)l, 0, 0);
#else
  unsigned lo = (unsigned)(size_t)(LDS_AS void*)l;
  unsigned long long ga = (unsigned long long)g;
  asm volatile("global_load_async_to_lds_b128 %0, %1, off"
               :: "v"(lo), "v"(ga) : "memory");
#endif
}

static __device__ __forceinline__ void wait_async0() {
#if __has_builtin(__builtin_amdgcn_s_wait_asynccnt)
  __builtin_amdgcn_s_wait_asynccnt(0);
#else
  asm volatile("s_wait_asynccnt 0x0" ::: "memory");
#endif
}

// ---------------------------------------------------------------------------
// Zero-padded f32 -> bf16 weight conversion: dst is (prows x pcols) row-major,
// src is (rows x cols); out-of-range entries are written as 0.
// ---------------------------------------------------------------------------
__global__ __launch_bounds__(256) void convert_pad_bf16(
    const float* __restrict__ src, unsigned short* __restrict__ dst,
    int rows, int cols, int prows, int pcols) {
  int idx = blockIdx.x * 256 + threadIdx.x;
  if (idx >= prows * pcols) return;
  int r = idx / pcols, c = idx - r * pcols;
  unsigned short v = 0;
  if (r < rows && c < cols) v = f2bf(src[(size_t)r * cols + c]);
  dst[idx] = v;
}

// ---------------------------------------------------------------------------
// Tiled WMMA GEMM:  C[m][n] = act( sum_k A[m][k] * B[n][k] + bias[n] )
//   A: (M x K) row-major, f32 (converted inline, VGPR-pipelined) or bf16
//      (async-to-LDS pipelined).
//   B: (Npad x K) row-major bf16, async-to-LDS pipelined.
//   Block tile 128x64, BK=64, 256 threads = 8 waves, wave tile 32x32 (2x2
//   WMMA), double-buffered LDS (stage s+1 in flight while stage s computes).
//   blockIdx.x = N tile (fast), blockIdx.y = M tile (slow) -> A-panel L2 reuse.
// ACT: 0 = identity, 1 = relu, 2 = sigmoid.  OT: float or ushort(bf16).
// ---------------------------------------------------------------------------
template <typename AT, typename OT, int ACT>
__global__ __launch_bounds__(256) void gemm_wmma_bf16(
    const AT* __restrict__ A, int lda,
    const unsigned short* __restrict__ Bw, int ldb,
    const float* __restrict__ bias, int biasN,
    OT* __restrict__ C, int ldc, int K) {
  __shared__ unsigned short As[2][128 * LDT];  // 2 x 18 KB
  __shared__ unsigned short Bs[2][64 * LDT];   // 2 x  9 KB

  const int tid   = threadIdx.x;
  const int lane  = tid & 31;
  const int wave  = tid >> 5;
  const int waveM = wave & 3;   // 4 waves along M (32 rows each)
  const int waveN = wave >> 2;  // 2 waves along N (32 cols each)

  const int rowBlk = blockIdx.y * 128;  // M (slow grid dim)
  const int colBlk = blockIdx.x * 64;   // N (fast grid dim)

  // cooperative tile-load mapping
  const int ar = tid >> 1;        // A: 128 rows, 32 halves per thread
  const int ac = (tid & 1) * 32;
  const int br = tid >> 2;        // B: 64 rows, 16 halves per thread
  const int bc = (tid & 3) * 16;

  // WMMA fragment lane mapping (ISA 7.12.2, 16-bit operands, wave32)
  const int halfSel = lane >> 4;       // 0: lanes 0-15, 1: lanes 16-31
  const int fr      = lane & 15;       // A row / B col / C col within tile
  const int akb     = halfSel * 8;     // A k-base inside 32-k step
  const int bkb     = halfSel * 8;     // B k-base inside 32-k step

  const int S = K / 64;  // number of 64-k stages (all K are multiples of 64)

  // ---- pipeline helpers -------------------------------------------------
  auto issueB = [&](int s, int p) {
    const unsigned short* src = Bw + (size_t)(colBlk + br) * ldb + (s * 64 + bc);
    unsigned short* dst = &Bs[p][br * LDT + bc];
#pragma unroll
    for (int i = 0; i < 16; i += 8) async_b128(src + i, dst + i);
  };
  auto issueA = [&](int s, int p) {  // bf16-A path only
    const unsigned short* src =
        (const unsigned short*)A + (size_t)(rowBlk + ar) * lda + (s * 64 + ac);
    unsigned short* dst = &As[p][ar * LDT + ac];
#pragma unroll
    for (int i = 0; i < 32; i += 8) async_b128(src + i, dst + i);
  };
  float areg[32];                    // f32-A path staging registers
  auto loadA = [&](int s) {
    const float* src =
        (const float*)A + (size_t)(rowBlk + ar) * lda + (s * 64 + ac);
#pragma unroll
    for (int i = 0; i < 32; i += 4)
      *reinterpret_cast<float4*>(&areg[i]) =
          *reinterpret_cast<const float4*>(src + i);
  };
  auto storeA = [&](int p) {         // pair-convert + publish to LDS
    unsigned short* dst = &As[p][ar * LDT + ac];
#pragma unroll
    for (int i = 0; i < 32; i += 8) {
      uint4 v;
      v.x = packbf2(areg[i + 0], areg[i + 1]);
      v.y = packbf2(areg[i + 2], areg[i + 3]);
      v.z = packbf2(areg[i + 4], areg[i + 5]);
      v.w = packbf2(areg[i + 6], areg[i + 7]);
      *reinterpret_cast<uint4*>(dst + i) = v;
    }
  };

  const v8f vzero = {0.f, 0.f, 0.f, 0.f, 0.f, 0.f, 0.f, 0.f};
  v8f acc[2][2];
#pragma unroll
  for (int i = 0; i < 2; ++i)
#pragma unroll
    for (int j = 0; j < 2; ++j) acc[i][j] = vzero;

  // ---- prologue: fill stage 0 -------------------------------------------
  if constexpr (sizeof(AT) == 4) {
    loadA(0);
    storeA(0);
  } else {
    issueA(0, 0);
  }
  issueB(0, 0);
  wait_async0();
  __syncthreads();  // compiler adds s_wait_dscnt before the barrier (f32 path)

  // ---- main pipelined loop ----------------------------------------------
  for (int s = 0; s < S; ++s) {
    const int p = s & 1;
    const bool more = (s + 1 < S);

    if (more) {
      if constexpr (sizeof(AT) == 4) loadA(s + 1);  // global->VGPR, overlaps
      else issueA(s + 1, p ^ 1);                    // async, overlaps compute
      issueB(s + 1, p ^ 1);
    }

    // ---- compute stage s from buffer p ----
    const unsigned short* Ab = As[p];
    const unsigned short* Bb = Bs[p];
#pragma unroll
    for (int kk = 0; kk < 2; ++kk) {  // two 32-k WMMA steps per stage
      v16bf afrag[2], bfrag[2];
#pragma unroll
      for (int tm = 0; tm < 2; ++tm) {
        const unsigned short* pp =
            &Ab[(waveM * 32 + tm * 16 + fr) * LDT + kk * 32 + akb];
        v8bf lo = *reinterpret_cast<const v8bf*>(pp);       // K kb..kb+7
        v8bf hi = *reinterpret_cast<const v8bf*>(pp + 16);  // K kb+16..kb+23
#pragma unroll
        for (int i = 0; i < 8; ++i) {
          afrag[tm][i] = lo[i];
          afrag[tm][8 + i] = hi[i];
        }
      }
#pragma unroll
      for (int tn = 0; tn < 2; ++tn) {
        const unsigned short* pp =
            &Bb[(waveN * 32 + tn * 16 + fr) * LDT + kk * 32 + bkb];
        v8bf lo = *reinterpret_cast<const v8bf*>(pp);      // K koff..koff+7
        v8bf hi = *reinterpret_cast<const v8bf*>(pp + 16); // K koff+16..+23
#pragma unroll
        for (int i = 0; i < 8; ++i) {
          bfrag[tn][i] = lo[i];
          bfrag[tn][8 + i] = hi[i];
        }
      }
#pragma unroll
      for (int tm = 0; tm < 2; ++tm)
#pragma unroll
        for (int tn = 0; tn < 2; ++tn)
          acc[tm][tn] = __builtin_amdgcn_wmma_f32_16x16x32_bf16(
              false, afrag[tm], false, bfrag[tn], (short)0, acc[tm][tn],
              false, false);
    }

    if (more) {
      if constexpr (sizeof(AT) == 4) storeA(p ^ 1);  // after reads of buf p
      wait_async0();   // stage s+1 async copies landed (own-wave, in-order)
      __syncthreads(); // all waves' copies landed + buf p reads retired
    }
  }

  // ---- epilogue: bias + activation + store ------------------------------
#pragma unroll
  for (int tn = 0; tn < 2; ++tn) {
    const int n = colBlk + waveN * 32 + tn * 16 + fr;
    const float bv = (n < biasN) ? bias[n] : 0.0f;
#pragma unroll
    for (int tm = 0; tm < 2; ++tm) {
      const int r0 = rowBlk + waveM * 32 + tm * 16 + halfSel * 8;
#pragma unroll
      for (int v = 0; v < 8; ++v) {
        float val = acc[tm][tn][v] + bv;
        if constexpr (ACT == 1) val = fmaxf(val, 0.0f);
        if constexpr (ACT == 2) val = 1.0f / (1.0f + __expf(-val));
        if constexpr (sizeof(OT) == 2)
          C[(size_t)(r0 + v) * ldc + n] = (OT)f2bf(val);
        else
          C[(size_t)(r0 + v) * ldc + n] = (OT)val;
      }
    }
  }
}

// ---------------------------------------------------------------------------
// Vector quantize: for each (b,d) row of z_e (B,K=128,D=8), argmin over 128
// codes of ||z - w_n||^2, then gather w_idx into emb_o (f32) and zq (bf16).
// One block = 8 samples (8 waves, one sample each); codebook cached in LDS.
// ---------------------------------------------------------------------------
__global__ __launch_bounds__(256) void vq_quantize(
    const float* __restrict__ z, const float* __restrict__ Wemb,
    float* __restrict__ embo, unsigned short* __restrict__ zq) {
  __shared__ float Wl[128 * 128];  // 64 KB, layout matches global: [k][n]
  __shared__ float wsqs[128];      // ||w_n||^2
  __shared__ float zbuf[8][128];   // per-wave z row

  const int tid = threadIdx.x, lane = tid & 31, w = tid >> 5;

  for (int i = tid; i < 128 * 128; i += 256) Wl[i] = Wemb[i];
  __syncthreads();
  if (tid < 128) {
    float s = 0.f;
    for (int k = 0; k < 128; ++k) {
      float t = Wl[k * 128 + tid];
      s += t * t;
    }
    wsqs[tid] = s;
  }
  __syncthreads();

  const int b = blockIdx.x * 8 + w;  // sample handled by this wave

  for (int d = 0; d < 8; ++d) {
    // stage z row (stride-8 gather from z_e) into wave-private LDS
#pragma unroll
    for (int j = 0; j < 4; ++j) {
      int k = lane + 32 * j;
      zbuf[w][k] = z[(size_t)b * 1024 + (size_t)k * 8 + d];
    }
    // wave-local visibility: lanes read each other's LDS stores (lockstep wave)
    asm volatile("s_wait_dscnt 0" ::: "memory");

    float dot0 = 0.f, dot1 = 0.f, dot2 = 0.f, dot3 = 0.f;
    for (int k = 0; k < 128; ++k) {
      float zk = zbuf[w][k];
      const float* wr = &Wl[k * 128 + lane];
      dot0 = fmaf(zk, wr[0], dot0);
      dot1 = fmaf(zk, wr[32], dot1);
      dot2 = fmaf(zk, wr[64], dot2);
      dot3 = fmaf(zk, wr[96], dot3);
    }
    // score = ||w||^2 - 2 z.w  (||z||^2 constant over n); first-min tie-break
    float best = wsqs[lane] - 2.f * dot0;
    int bi = lane;
    float c;
    c = wsqs[lane + 32] - 2.f * dot1; if (c < best) { best = c; bi = lane + 32; }
    c = wsqs[lane + 64] - 2.f * dot2; if (c < best) { best = c; bi = lane + 64; }
    c = wsqs[lane + 96] - 2.f * dot3; if (c < best) { best = c; bi = lane + 96; }
#pragma unroll
    for (int off = 16; off > 0; off >>= 1) {
      float ov = __shfl_xor(best, off, 32);
      int oi = __shfl_xor(bi, off, 32);
      if (ov < best || (ov == best && oi < bi)) { best = ov; bi = oi; }
    }
    // gather selected code column and scatter to both outputs
#pragma unroll
    for (int j = 0; j < 4; ++j) {
      int k = lane + 32 * j;
      float v = Wl[k * 128 + bi];
      size_t o = (size_t)b * 1024 + (size_t)k * 8 + d;
      embo[o] = v;
      zq[o] = f2bf(v);
    }
  }
}

// ---------------------------------------------------------------------------
// Host-side orchestration (all launches on `stream`; graph-capture safe).
// ---------------------------------------------------------------------------
extern "C" void kernel_launch(void* const* d_in, const int* in_sizes, int n_in,
                              void* d_out, int out_size, void* d_ws,
                              size_t ws_size, hipStream_t stream) {
  (void)in_sizes; (void)n_in; (void)out_size; (void)ws_size;
  const float* x    = (const float*)d_in[0];
  const float* W1   = (const float*)d_in[1];
  const float* b1   = (const float*)d_in[2];
  const float* W2   = (const float*)d_in[3];
  const float* b2   = (const float*)d_in[4];
  const float* W3   = (const float*)d_in[5];
  const float* b3   = (const float*)d_in[6];
  const float* W4   = (const float*)d_in[7];
  const float* b4   = (const float*)d_in[8];
  const float* Wemb = (const float*)d_in[9];

  float* recon = (float*)d_out;                     // 16384 x 3072
  float* z_e   = recon + (size_t)16384 * 3072;      // 16384 x 1024
  float* emb_o = z_e + (size_t)16384 * 1024;        // 16384 x 1024 (B,K,D flat)

  // Workspace layout (bytes); ~53 MB total, 256B-aligned chunks.
  char* ws = (char*)d_ws;
  unsigned short* W1p = (unsigned short*)(ws);             // 448 x 3072
  unsigned short* W2p = (unsigned short*)(ws +  2752512);  // 1024 x 448
  unsigned short* W3p = (unsigned short*)(ws +  3670016);  // 448 x 1024
  unsigned short* W4p = (unsigned short*)(ws +  4587520);  // 3072 x 448
  unsigned short* h   = (unsigned short*)(ws +  7340032);  // 16384 x 448 (h1/h3)
  unsigned short* zq  = (unsigned short*)(ws + 22020096);  // 16384 x 1024

  dim3 blk(256);

  // Pad + convert weights to bf16 (zero-filled pad rows/cols).
  convert_pad_bf16<<<(448 * 3072 + 255) / 256, blk, 0, stream>>>(W1, W1p, 400, 3072, 448, 3072);
  convert_pad_bf16<<<(1024 * 448 + 255) / 256, blk, 0, stream>>>(W2, W2p, 1024, 400, 1024, 448);
  convert_pad_bf16<<<(448 * 1024 + 255) / 256, blk, 0, stream>>>(W3, W3p, 400, 1024, 448, 1024);
  convert_pad_bf16<<<(3072 * 448 + 255) / 256, blk, 0, stream>>>(W4, W4p, 3072, 400, 3072, 448);

  // Grid = (Ntiles, Mtiles): x fastest -> all N-blocks of one A panel run
  // consecutively, keeping the 1.5 MB panel in L2.
  // h1 = relu(x @ W1^T + b1)              M=16384 N=448(pad of 400) K=3072
  gemm_wmma_bf16<float, unsigned short, 1><<<dim3(7, 128), blk, 0, stream>>>(
      x, 3072, W1p, 3072, b1, 400, h, 448, 3072);
  // z_e = h1 @ W2^T + b2                  M=16384 N=1024 K=448(pad of 400)
  gemm_wmma_bf16<unsigned short, float, 0><<<dim3(16, 128), blk, 0, stream>>>(
      h, 448, W2p, 448, b2, 1024, z_e, 1024, 448);
  // VQ: emb_o (f32, d_out) and z_q (bf16, ws)
  vq_quantize<<<16384 / 8, blk, 0, stream>>>(z_e, Wemb, emb_o, zq);
  // h3 = relu(z_q @ W3^T + b3)            M=16384 N=448(pad) K=1024
  gemm_wmma_bf16<unsigned short, unsigned short, 1><<<dim3(7, 128), blk, 0, stream>>>(
      zq, 1024, W3p, 1024, b3, 400, h, 448, 1024);
  // recon = sigmoid(h3 @ W4^T + b4)       M=16384 N=3072 K=448(pad)
  gemm_wmma_bf16<unsigned short, float, 2><<<dim3(48, 128), blk, 0, stream>>>(
      h, 448, W4p, 448, b4, 3072, recon, 3072, 448);
}
